// Transition_LSTM_72249939853922
// MI455X (gfx1250) — matrix operand order repaired and checked by reading, not compiled, not run on hardware
//
#include <hip/hip_runtime.h>
#include <hip/hip_bf16.h>

// ---------------------------------------------------------------------------
// LSTM recurrence, T serial steps, batch-parallel across workgroups.
// 32 wave32s per workgroup, 32 batch rows per workgroup: each wave owns one
// 16-wide hid column group for all four gates and computes TWO 16-row M-tiles
// against the same B fragments -> 1 global b128 load per WMMA (~32 FLOP/B vs
// L2). B loads use uniform-base + lane-offset + immediate (GVS form) so the
// inner loop has no vector address arithmetic (no WMMA->VALU hazard NOPs).
// Mixed precision: v_wmma_f32_16x16x32_f16 with f32 accumulation.
// Cell state c: two v8f fragments per wave, in registers for all T steps.
// LDS [x|h] state is double-buffered: h_new/z_new write the next buffer.
// Activations: v_exp_f32 + v_rcp_f32 (no IEEE divide expansion).
// ---------------------------------------------------------------------------

typedef __attribute__((ext_vector_type(16))) _Float16 v16h_t;
typedef __attribute__((ext_vector_type(8)))  _Float16 v8h_t;
typedef __attribute__((ext_vector_type(8)))  float    v8f_t;

#define HID    512
#define LAT    64
#define NGATE  2048            // 4*HID
#define KCAT   576             // LAT + HID (concatenated K for the gate GEMM)
#define PITCH  584             // LDS row pitch in halves (mult. of 8 -> 16B-aligned b128)
#define ROWS   32              // batch rows per workgroup (two 16-row M-tiles)
#define BATCH  4096

// Workspace layout (bytes)
#define WC_BYTES    (NGATE * KCAT * 2)            // 2,359,296  f16 [2048][576]
#define BCAT_OFF    WC_BYTES
#define BCAT_BYTES  (NGATE * 4)                   // 8,192      f32 [2048]
#define WOUT_OFF    (BCAT_OFF + BCAT_BYTES)
#define WOUT_BYTES  (LAT * HID * 2)               // 65,536     f16 [64][512]

// ----- helpers --------------------------------------------------------------

__device__ __forceinline__ float fast_rcp(float x) {
    return __builtin_amdgcn_rcpf(x);            // v_rcp_f32 (TRANS32)
}

__device__ __forceinline__ float sigf(float x) {
    return fast_rcp(1.0f + __expf(-x));
}

__device__ __forceinline__ float tanh_fast(float x) {
    // tanh(x) = 1 - 2/(exp(2x)+1); clamp so exp stays finite
    float xc = fminf(fmaxf(x, -15.0f), 15.0f);
    float e  = __expf(2.0f * xc);
    return 1.0f - 2.0f * fast_rcp(e + 1.0f);
}

__device__ __forceinline__ v16h_t cat16(v8h_t lo, v8h_t hi) {
    return __builtin_shufflevector(lo, hi, 0,1,2,3,4,5,6,7,8,9,10,11,12,13,14,15);
}

// A-fragment (16x32 f16, MxK) from LDS rows (pitch PITCH halves).
// Lane L: M = L&15 (at rowbase). Halves: K = {kb+hi*16..+15}.
__device__ __forceinline__ v16h_t load_a_frag(const _Float16* xh, int row, int kb, int hi) {
    const _Float16* p = xh + row * PITCH + kb + hi * 16;
    v8h_t lo = *(const v8h_t*)p;
    v8h_t h2 = *(const v8h_t*)(p + 8);
    return cat16(lo, h2);
}

// B-fragment (32x16 f16, KxN): uniform base pointer + (lane-variant offset +
// compile-time-constant offset) -> backend selects saddr + voffset + imm form.
__device__ __forceinline__ v16h_t load_b_off(const _Float16* base, int off) {
    v8h_t lo = *(const v8h_t*)(base + off);
    v8h_t h2 = *(const v8h_t*)(base + off + 8);
    return cat16(lo, h2);
}

// ----- prep kernel: pack weights to f16, fuse biases ------------------------

__global__ __launch_bounds__(256) void lstm_prep_kernel(
    const float* __restrict__ Wih,   // [2048][64]
    const float* __restrict__ Whh,   // [2048][512]
    const float* __restrict__ bih,   // [2048]
    const float* __restrict__ bhh,   // [2048]
    const float* __restrict__ Wout,  // [64][512]
    _Float16* __restrict__ Wc,       // [2048][576] f16
    float*    __restrict__ bcat,     // [2048]
    _Float16* __restrict__ WoutH)    // [64][512] f16
{
    int idx = blockIdx.x * 256 + threadIdx.x;
    if (idx < NGATE * KCAT) {
        int n = idx / KCAT;
        int k = idx - n * KCAT;
        float v = (k < LAT) ? Wih[n * LAT + k] : Whh[n * HID + (k - LAT)];
        Wc[idx] = (_Float16)v;
    }
    if (idx < NGATE) bcat[idx] = bih[idx] + bhh[idx];
    if (idx < LAT * HID) WoutH[idx] = (_Float16)Wout[idx];
}

// ----- main kernel: 1024 threads = 32 waves; 32 batch rows for all T steps --

__global__ __launch_bounds__(1024) void lstm_recur_kernel(
    const int*   __restrict__ Tp,
    const float* __restrict__ z0,     // [4096][64]
    const _Float16* __restrict__ Wc,  // [2048][576]
    const float* __restrict__ bcat,   // [2048]
    const _Float16* __restrict__ Wo,  // [64][512]
    const float* __restrict__ bo,     // [64]
    float* __restrict__ out)          // [T*4096][64]
{
    __shared__ _Float16 xh[2][ROWS * PITCH];   // [x(0..63) | h(64..575)] per row, f16

    const int tid  = threadIdx.x;
    const int lane = tid & 31;
    const int wave = tid >> 5;          // 0..31, owns hid cols [wave*16, wave*16+16)
    const int hi   = (lane >> 4) & 1;   // 0: lanes 0-15, 1: lanes 16-31
    const int ln   = lane & 15;
    const int row0 = blockIdx.x * ROWS;
    const int T    = *Tp;

    // init buffer 0: x = z0 (f16), h = 0
    for (int idx = tid; idx < ROWS * PITCH; idx += 1024) {
        int m = idx / PITCH;
        int k = idx - m * PITCH;
        float v = (k < LAT) ? z0[(row0 + m) * LAT + k] : 0.0f;
        xh[0][idx] = (_Float16)v;
    }

    // persistent cell state, D-fragment layout, one fragment per M-tile:
    //   c0[r] = c(row = r + hi*8,      col = wave*16 + ln)
    //   c1[r] = c(row = 16 + r + hi*8, col = wave*16 + ln)
    v8f_t c0, c1;
#pragma unroll
    for (int r = 0; r < 8; ++r) { c0[r] = 0.0f; c1[r] = 0.0f; }

    // hoisted per-gate biases (constant across steps)
    const float gb0 = bcat[0 * HID + wave * 16 + ln];
    const float gb1 = bcat[1 * HID + wave * 16 + ln];
    const float gb2 = bcat[2 * HID + wave * 16 + ln];
    const float gb3 = bcat[3 * HID + wave * 16 + ln];

    // loop-invariant per-lane element offsets (gate deltas become immediates)
    const int bvoff = (wave * 16 + ln) * KCAT + hi * 16;        // into Wc
    const int zvoff = ((wave & 3) * 16 + ln) * HID + hi * 16;   // into Wo
    const float zb  = (wave < 8) ? bo[(wave & 3) * 16 + ln] : 0.0f;

    __syncthreads();

    int cur = 0;
    for (int t = 0; t < T; ++t) {
        const _Float16* xc = xh[cur];        // read buffer
        _Float16*       xn = xh[cur ^ 1];    // write buffer (h_new, z_new)

        // ---- gate GEMM: 8 independent accumulation chains (2 M-tiles x 4 gates),
        //      each B fragment feeds two WMMAs (A-tile 0 and A-tile 1).
        v8f_t a00, a01, a02, a03, a10, a11, a12, a13;
#pragma unroll
        for (int r = 0; r < 8; ++r) {
            a00[r] = gb0; a01[r] = gb1; a02[r] = gb2; a03[r] = gb3;
            a10[r] = gb0; a11[r] = gb1; a12[r] = gb2; a13[r] = gb3;
        }

#pragma unroll 1
        for (int kt = 0; kt < KCAT / 32; ++kt) {
            const int kb = kt * 32;
            const _Float16* bk = Wc + kb;        // uniform (SGPR) base, advances per kt
            v16h_t fa0 = load_a_frag(xc, ln,      kb, hi);
            v16h_t fa1 = load_a_frag(xc, 16 + ln, kb, hi);
            v16h_t b0 = load_b_off(bk, bvoff + 0 * HID * KCAT);
            v16h_t b1 = load_b_off(bk, bvoff + 1 * HID * KCAT);
            v16h_t b2 = load_b_off(bk, bvoff + 2 * HID * KCAT);
            v16h_t b3 = load_b_off(bk, bvoff + 3 * HID * KCAT);
            a00 = __builtin_amdgcn_wmma_f32_16x16x32_f16(false, fa0, false, b0, (short)0, a00, false, false);
            a10 = __builtin_amdgcn_wmma_f32_16x16x32_f16(false, fa1, false, b0, (short)0, a10, false, false);
            a01 = __builtin_amdgcn_wmma_f32_16x16x32_f16(false, fa0, false, b1, (short)0, a01, false, false);
            a11 = __builtin_amdgcn_wmma_f32_16x16x32_f16(false, fa1, false, b1, (short)0, a11, false, false);
            a02 = __builtin_amdgcn_wmma_f32_16x16x32_f16(false, fa0, false, b2, (short)0, a02, false, false);
            a12 = __builtin_amdgcn_wmma_f32_16x16x32_f16(false, fa1, false, b2, (short)0, a12, false, false);
            a03 = __builtin_amdgcn_wmma_f32_16x16x32_f16(false, fa0, false, b3, (short)0, a03, false, false);
            a13 = __builtin_amdgcn_wmma_f32_16x16x32_f16(false, fa1, false, b3, (short)0, a13, false, false);
        }

        // ---- nonlinearities + cell update in registers; h_new -> next buffer
        {
            const int j = LAT + wave * 16 + ln;     // h column in [64, 576)
#pragma unroll
            for (int r = 0; r < 8; ++r) {
                // M-tile 0 (rows 0..15)
                float iv = sigf(a00[r]);
                float fv = sigf(a01[r]);
                float gv = tanh_fast(a02[r]);
                float ov = sigf(a03[r]);
                float cn = fv * c0[r] + iv * gv;
                c0[r] = cn;
                xn[(r + hi * 8) * PITCH + j] = (_Float16)(ov * tanh_fast(cn));
                // M-tile 1 (rows 16..31)
                iv = sigf(a10[r]);
                fv = sigf(a11[r]);
                gv = tanh_fast(a12[r]);
                ov = sigf(a13[r]);
                cn = fv * c1[r] + iv * gv;
                c1[r] = cn;
                xn[(16 + r + hi * 8) * PITCH + j] = (_Float16)(ov * tanh_fast(cn));
            }
        }

        __syncthreads();   // h_new complete in next buffer

        // ---- z = h @ Wout^T + b_out : 2 M-tiles x 4 N-tiles -> waves 0..7
        if (wave < 8) {
            const int mt = wave >> 2;       // 0 or 1: M-tile
            const int nt = wave & 3;        // 0..3 : N-tile
            v8f_t z;
#pragma unroll
            for (int r = 0; r < 8; ++r) z[r] = zb;
#pragma unroll 1
            for (int kt = 0; kt < HID / 32; ++kt) {
                v16h_t a = load_a_frag(xn, mt * 16 + ln, LAT + kt * 32, hi);
                v16h_t b = load_b_off(Wo + kt * 32, zvoff);
                z = __builtin_amdgcn_wmma_f32_16x16x32_f16(false, a, false, b, (short)0, z, false, false);
            }
            const int n = nt * 16 + ln;
#pragma unroll
            for (int r = 0; r < 8; ++r) {
                const int m = mt * 16 + r + hi * 8;
                xn[m * PITCH + n] = (_Float16)z[r];                     // next step's x
                out[((size_t)t * BATCH + row0 + m) * LAT + n] = z[r];   // step output
            }
        }

        __syncthreads();   // next buffer fully ready
        cur ^= 1;
    }
}

// ----- launcher -------------------------------------------------------------

extern "C" void kernel_launch(void* const* d_in, const int* in_sizes, int n_in,
                              void* d_out, int out_size, void* d_ws, size_t ws_size,
                              hipStream_t stream) {
    (void)in_sizes; (void)n_in; (void)out_size; (void)ws_size;

    const int*   Tp   = (const int*)  d_in[0];
    const float* z0   = (const float*)d_in[1];
    const float* Wih  = (const float*)d_in[2];
    const float* Whh  = (const float*)d_in[3];
    const float* bih  = (const float*)d_in[4];
    const float* bhh  = (const float*)d_in[5];
    const float* Wout = (const float*)d_in[6];
    const float* bout = (const float*)d_in[7];

    _Float16* Wc    = (_Float16*)d_ws;
    float*    bcat  = (float*)   ((char*)d_ws + BCAT_OFF);
    _Float16* WoutH = (_Float16*)((char*)d_ws + WOUT_OFF);

    // pack weights to f16 (deterministic, inputs untouched)
    int prep_elems  = NGATE * KCAT;
    int prep_blocks = (prep_elems + 255) / 256;
    lstm_prep_kernel<<<prep_blocks, 256, 0, stream>>>(Wih, Whh, bih, bhh, Wout,
                                                      Wc, bcat, WoutH);

    // recurrence: 128 workgroups x 32 batch rows, 32 wave32s each
    lstm_recur_kernel<<<BATCH / ROWS, 1024, 0, stream>>>(Tp, z0, Wc, bcat, WoutH,
                                                         bout, (float*)d_out);
}
